// NexusGraph_89867895701535
// MI455X (gfx1250) — matrix operand order repaired and checked by reading, not compiled, not run on hardware
//
#include <hip/hip_runtime.h>
#include <hip/hip_bf16.h>
#include <math.h>

#define N_NODES 50000
#define N_EDGES 800000
#define EDGE_DIM 32

typedef __attribute__((ext_vector_type(16))) __bf16 v16bf;
typedef __attribute__((ext_vector_type(8)))  float  v8f;

// ---------- bf16 helpers (bit-level, round-to-nearest-even) ----------
__device__ __forceinline__ unsigned short f2bf(float f) {
    unsigned int u = __float_as_uint(f);
    u += 0x7FFFu + ((u >> 16) & 1u);
    return (unsigned short)(u >> 16);
}
__device__ __forceinline__ float bf2f(unsigned short b) {
    return __uint_as_float(((unsigned int)b) << 16);
}
__device__ __forceinline__ float bflo(unsigned int u) {   // low bf16 of a packed pair
    return __uint_as_float(u << 16);
}
__device__ __forceinline__ float bfhi(unsigned int u) {   // high bf16 of a packed pair
    return __uint_as_float(u & 0xFFFF0000u);
}

// float atomic max via signed/unsigned integer ordering trick
__device__ __forceinline__ void atomicMaxFloat(float* addr, float value) {
    if (value >= 0.0f) atomicMax((int*)addr, __float_as_int(value));
    else               atomicMin((unsigned int*)addr, __float_as_uint(value));
}

// ---------- elementwise conversion ----------
__global__ void k_f32_to_bf16(const float* __restrict__ in,
                              unsigned short* __restrict__ out, long n) {
    long i = (long)blockIdx.x * blockDim.x + threadIdx.x;
    if (i < n) out[i] = f2bf(in[i]);
}

__global__ void k_relu_bf16(const float* __restrict__ in,
                            unsigned short* __restrict__ out, long n) {
    long i = (long)blockIdx.x * blockDim.x + threadIdx.x;
    if (i < n) out[i] = f2bf(fmaxf(in[i], 0.0f));
}

__global__ void k_init_softmax(float* __restrict__ nmax, float* __restrict__ nsum, int n) {
    int i = blockIdx.x * blockDim.x + threadIdx.x;
    if (i < n) { nmax[i] = -3.0e38f; nsum[i] = 0.0f; }
}

// ---------- pack fp32 weights [K,Nc] into WMMA B-fragment order ----------
// Layout: Wp[kt][nt][lane][el], kt=k/32, nt=n/16, lane=(k%32/16)*16 + n%16, el=k%16.
// Each lane's B fragment is then 32 contiguous bytes at (tile*512 + lane*16).
__global__ void k_pack_w(const float* __restrict__ W, unsigned short* __restrict__ Wp,
                         int K, int Nc) {
    int i = blockIdx.x * blockDim.x + threadIdx.x;
    if (i >= K * Nc) return;
    int k = i / Nc, n = i - k * Nc;
    int kt = k >> 5, kin = k & 31;
    int nt = n >> 4, nin = n & 15;
    int lane = ((kin >> 4) << 4) | nin;
    int el   = kin & 15;
    Wp[(((size_t)kt * (Nc >> 4) + nt) << 9) + (lane << 4) + el] = f2bf(W[i]);
}

// ---------- WMMA GEMM: C[M,Nc] = A[M,K](bf16) @ B[K,Nc](packed bf16) + bias ----------
// 8 waves per block; each wave computes NT adjacent 16x16 tiles (one A fragment is
// reused across NT v_wmma_f32_16x16x32_bf16 ops). A is staged through a per-wave LDS
// slice; B fragments are loaded directly from the packed weight buffer.
template <int NT, bool F32OUT>
__global__ __launch_bounds__(256) void k_wmma_gemm(
    const unsigned short* __restrict__ A,    // [M,K] bf16 row-major
    const unsigned short* __restrict__ Bp,   // packed fragments [K/32][Nc/16][32][16]
    const float* __restrict__ bias,          // nullable
    unsigned short* __restrict__ outb,       // bf16 out (NT path F32OUT=false)
    float* __restrict__ outf,                // f32 out  (F32OUT=true)
    int M, int K, int Nc)
{
    __shared__ __align__(16) unsigned short sA[8][512];   // 16x32 bf16 per wave
    const int lane = threadIdx.x & 31;
    const int wave = threadIdx.x >> 5;
    const int tilesN  = Nc >> 4;
    const int groupsN = tilesN / NT;
    const int tasks   = (M >> 4) * groupsN;
    const int tid = blockIdx.x * 8 + wave;
    if (tid >= tasks) return;
    const int m0  = (tid / groupsN) << 4;
    const int nt0 = (tid % groupsN) * NT;
    unsigned short* sAw = sA[wave];

    const int nloc = lane & 15;
    const int half = lane >> 4;
    v8f acc[NT];
    #pragma unroll
    for (int t = 0; t < NT; ++t) acc[t] = (v8f){};

    for (int k0 = 0; k0 < K; k0 += 32) {
        // stage A tile 16x32 bf16 = 64 x 16B chunks, 2 per lane
        #pragma unroll
        for (int t2 = 0; t2 < 2; ++t2) {
            int c = lane + 32 * t2;
            int row  = c >> 2;
            int col8 = (c & 3) << 3;
            ((uint4*)sAw)[c] = *(const uint4*)(A + (size_t)(m0 + row) * K + k0 + col8);
        }
        // prefetch next k-slab of A while we consume this one
        if (k0 + 32 < K) {
            int row = lane >> 2, col8 = (lane & 3) << 3;
            __builtin_prefetch(A + (size_t)(m0 + row) * K + k0 + 32 + col8, 0, 3);
        }
        // DS ops are in-order per wave; wait for tile stores to land before reads
        asm volatile("s_wait_dscnt 0" ::: "memory");

        const __bf16* lA = (const __bf16*)sAw;
        v16bf af;
        // A fragment: k = 16*half + element (contiguous 16-element run, matches packed B)
        #pragma unroll
        for (int i = 0; i < 8; ++i) {
            int kk = (half << 4) + (i << 1);
            af[2 * i]     = lA[nloc * 32 + kk];
            af[2 * i + 1] = lA[nloc * 32 + kk + 1];
        }
        const size_t kt = (size_t)(k0 >> 5);
        #pragma unroll
        for (int t = 0; t < NT; ++t) {
            v16bf bfrag = *(const v16bf*)(Bp + ((kt * tilesN + nt0 + t) << 9) + (lane << 4));
            acc[t] = __builtin_amdgcn_wmma_f32_16x16x32_bf16(false, af, false, bfrag,
                                                             (short)0, acc[t], false, false);
        }
    }

    // C/D layout: VGPR j -> row m0 + j + 8*half, col n0 + lane%16
    #pragma unroll
    for (int t = 0; t < NT; ++t) {
        const int n0 = (nt0 + t) << 4;
        const float bval = bias ? bias[n0 + nloc] : 0.0f;
        #pragma unroll
        for (int j = 0; j < 8; ++j) {
            int r = m0 + j + (half << 3);
            size_t o = (size_t)r * Nc + n0 + nloc;
            float v = acc[t][j] + bval;
            if (F32OUT) outf[o] = v;
            else        outb[o] = f2bf(v);
        }
    }
}

// ---------- edge attention: alpha = <q[dst], k[src]+e> * scale ; segment max ----------
__global__ __launch_bounds__(256) void k_edge_alpha(
    const int* __restrict__ src, const int* __restrict__ dst,
    const unsigned short* __restrict__ q, const unsigned short* __restrict__ k,
    const unsigned short* __restrict__ e, float* __restrict__ alpha,
    float* __restrict__ nmax, int d, float scale)
{
    int edge = blockIdx.x * 8 + (threadIdx.x >> 5);
    if (edge >= N_EDGES) return;
    int lane = threadIdx.x & 31;
    int s = src[edge], t = dst[edge];
    const unsigned int* qr = (const unsigned int*)(q + (size_t)t * d);
    const unsigned int* kr = (const unsigned int*)(k + (size_t)s * d);
    const unsigned int* er = (const unsigned int*)(e + (size_t)edge * d);
    const int dp = d >> 1;          // packed bf16 pairs
    float sum = 0.0f;
    for (int c = lane; c < dp; c += 32) {
        unsigned int qv = qr[c], kv = kr[c], ev = er[c];
        sum += bflo(qv) * (bflo(kv) + bflo(ev));
        sum += bfhi(qv) * (bfhi(kv) + bfhi(ev));
    }
    #pragma unroll
    for (int off = 16; off > 0; off >>= 1)
        sum += __shfl_down(sum, off, 32);
    if (lane == 0) {
        float a = sum * scale;
        alpha[edge] = a;
        atomicMaxFloat(&nmax[t], a);
    }
}

// ---------- exp(alpha - max) + segment sum (alpha overwritten in place) ----------
__global__ void k_edge_exp(const int* __restrict__ dst, float* __restrict__ alpha,
                           const float* __restrict__ nmax, float* __restrict__ nsum)
{
    int i = blockIdx.x * blockDim.x + threadIdx.x;
    if (i >= N_EDGES) return;
    int t = dst[i];
    float a = __expf(alpha[i] - nmax[t]);
    alpha[i] = a;
    atomicAdd(&nsum[t], a);
}

// ---------- agg[dst] += (a/sum[dst]) * (v[src] + e) ----------
__global__ __launch_bounds__(256) void k_edge_agg(
    const int* __restrict__ src, const int* __restrict__ dst,
    const unsigned short* __restrict__ v, const unsigned short* __restrict__ e,
    const float* __restrict__ a, const float* __restrict__ nsum,
    float* __restrict__ agg, int d)
{
    int edge = blockIdx.x * 8 + (threadIdx.x >> 5);
    if (edge >= N_EDGES) return;
    int lane = threadIdx.x & 31;
    int s = src[edge], t = dst[edge];
    float w = a[edge] / nsum[t];
    const unsigned int* vr = (const unsigned int*)(v + (size_t)s * d);
    const unsigned int* er = (const unsigned int*)(e + (size_t)edge * d);
    float* ar = agg + (size_t)t * d;
    const int dp = d >> 1;
    for (int c = lane; c < dp; c += 32) {
        unsigned int vv = vr[c], ev = er[c];
        atomicAdd(&ar[2 * c],     w * (bflo(vv) + bflo(ev)));
        atomicAdd(&ar[2 * c + 1], w * (bfhi(vv) + bfhi(ev)));
    }
}

// ---------- head: out[n] = <h3[n], Wc> + bc ----------
__global__ __launch_bounds__(256) void k_final(
    const unsigned short* __restrict__ h, const float* __restrict__ Wc,
    const float* __restrict__ bc, float* __restrict__ out)
{
    int node = blockIdx.x * 8 + (threadIdx.x >> 5);
    if (node >= N_NODES) return;
    int lane = threadIdx.x & 31;
    float s = bf2f(h[(size_t)node * 32 + lane]) * Wc[lane];
    #pragma unroll
    for (int off = 16; off > 0; off >>= 1) s += __shfl_down(s, off, 32);
    if (lane == 0) out[node] = s + bc[0];
}

extern "C" void kernel_launch(void* const* d_in, const int* in_sizes, int n_in,
                              void* d_out, int out_size, void* d_ws, size_t ws_size,
                              hipStream_t stream)
{
    (void)in_sizes; (void)n_in; (void)out_size; (void)ws_size;
    const float* x         = (const float*)d_in[0];
    const int*   ei        = (const int*)d_in[1];
    const float* edge_attr = (const float*)d_in[2];
    const int* src = ei;
    const int* dst = ei + N_EDGES;

    struct Layer { const float *Wq,*bq,*Wk,*bk,*Wv,*bv,*We,*Ws,*bs; int fi, fo; };
    Layer L[3];
    const int fis[3] = {128, 128, 64};
    const int fos[3] = {128, 64, 32};
    int idx = 3;
    for (int l = 0; l < 3; ++l) {
        L[l].Wq = (const float*)d_in[idx++]; L[l].bq = (const float*)d_in[idx++];
        L[l].Wk = (const float*)d_in[idx++]; L[l].bk = (const float*)d_in[idx++];
        L[l].Wv = (const float*)d_in[idx++]; L[l].bv = (const float*)d_in[idx++];
        L[l].We = (const float*)d_in[idx++];
        L[l].Ws = (const float*)d_in[idx++]; L[l].bs = (const float*)d_in[idx++];
        L[l].fi = fis[l]; L[l].fo = fos[l];
    }
    const float* Wc = (const float*)d_in[idx++];
    const float* bc = (const float*)d_in[idx++];
    float* out = (float*)d_out;

    // -------- workspace suballocation (~340 MB total) --------
    char* p = (char*)d_ws;
    auto alloc = [&](size_t bytes) -> char* {
        char* r = p; p += (bytes + 255) & ~(size_t)255; return r;
    };
    unsigned short* xb   = (unsigned short*)alloc((size_t)N_NODES * 128 * 2);      // activations (bf16)
    unsigned short* eb   = (unsigned short*)alloc((size_t)N_EDGES * EDGE_DIM * 2); // edge_attr (bf16)
    unsigned short* qb   = (unsigned short*)alloc((size_t)N_NODES * 128 * 2);
    unsigned short* kb   = (unsigned short*)alloc((size_t)N_NODES * 128 * 2);
    unsigned short* vb   = (unsigned short*)alloc((size_t)N_NODES * 128 * 2);
    unsigned short* ebl  = (unsigned short*)alloc((size_t)N_EDGES * 128 * 2);      // per-layer edge proj
    float*          agg  = (float*)alloc((size_t)N_NODES * 128 * 4);
    float*          alpha= (float*)alloc((size_t)N_EDGES * 4);
    float*          nmax = (float*)alloc((size_t)N_NODES * 4);
    float*          nsum = (float*)alloc((size_t)N_NODES * 4);
    unsigned short* wsc  = (unsigned short*)alloc((size_t)200000 * 2);             // packed bf16 weights

    auto pack = [&](const float* W, unsigned short* Wp, int K, int Nc) {
        int n = K * Nc;
        k_pack_w<<<(n + 255) / 256, 256, 0, stream>>>(W, Wp, K, Nc);
    };
    auto gemm = [&](const unsigned short* Aa, const unsigned short* Bp, const float* bias,
                    unsigned short* ob, float* of, int M, int K, int Nc) {
        int tilesN = Nc >> 4;
        if ((tilesN & 3) == 0) {
            int tasks = (M >> 4) * (tilesN >> 2);
            if (of) k_wmma_gemm<4, true ><<<(tasks + 7) / 8, 256, 0, stream>>>(Aa, Bp, bias, ob, of, M, K, Nc);
            else    k_wmma_gemm<4, false><<<(tasks + 7) / 8, 256, 0, stream>>>(Aa, Bp, bias, ob, of, M, K, Nc);
        } else {
            int tasks = (M >> 4) * (tilesN >> 1);
            if (of) k_wmma_gemm<2, true ><<<(tasks + 7) / 8, 256, 0, stream>>>(Aa, Bp, bias, ob, of, M, K, Nc);
            else    k_wmma_gemm<2, false><<<(tasks + 7) / 8, 256, 0, stream>>>(Aa, Bp, bias, ob, of, M, K, Nc);
        }
    };

    k_f32_to_bf16<<<(int)(((long)N_NODES * 128 + 255) / 256), 256, 0, stream>>>(
        x, xb, (long)N_NODES * 128);
    k_f32_to_bf16<<<(int)(((long)N_EDGES * EDGE_DIM + 255) / 256), 256, 0, stream>>>(
        edge_attr, eb, (long)N_EDGES * EDGE_DIM);

    for (int l = 0; l < 3; ++l) {
        const int fi = L[l].fi, fo = L[l].fo;
        unsigned short* Wqb = wsc;
        unsigned short* Wkb = Wqb + fi * fo;
        unsigned short* Wvb = Wkb + fi * fo;
        unsigned short* Wsb = Wvb + fi * fo;
        unsigned short* Web = Wsb + fi * fo;
        pack(L[l].Wq, Wqb, fi, fo);
        pack(L[l].Wk, Wkb, fi, fo);
        pack(L[l].Wv, Wvb, fi, fo);
        pack(L[l].Ws, Wsb, fi, fo);
        pack(L[l].We, Web, EDGE_DIM, fo);

        gemm(xb, Wqb, L[l].bq, qb, nullptr, N_NODES, fi, fo);
        gemm(xb, Wkb, L[l].bk, kb, nullptr, N_NODES, fi, fo);
        gemm(xb, Wvb, L[l].bv, vb, nullptr, N_NODES, fi, fo);
        gemm(xb, Wsb, L[l].bs, nullptr, agg, N_NODES, fi, fo);   // skip path seeds agg (f32)
        gemm(eb, Web, nullptr, ebl, nullptr, N_EDGES, EDGE_DIM, fo);

        k_init_softmax<<<(N_NODES + 255) / 256, 256, 0, stream>>>(nmax, nsum, N_NODES);
        const float scale = 1.0f / sqrtf((float)fo);
        k_edge_alpha<<<(N_EDGES + 7) / 8, 256, 0, stream>>>(src, dst, qb, kb, ebl,
                                                            alpha, nmax, fo, scale);
        k_edge_exp<<<(N_EDGES + 255) / 256, 256, 0, stream>>>(dst, alpha, nmax, nsum);
        k_edge_agg<<<(N_EDGES + 7) / 8, 256, 0, stream>>>(src, dst, vb, ebl, alpha,
                                                          nsum, agg, fo);
        k_relu_bf16<<<(int)(((long)N_NODES * fo + 255) / 256), 256, 0, stream>>>(
            agg, xb, (long)N_NODES * fo);
    }

    k_final<<<(N_NODES + 7) / 8, 256, 0, stream>>>(xb, Wc, bc, out);
}
